// GraphTransformer_69939247448732
// MI455X (gfx1250) — compile-verified, matrix operand order checked by reading, not run on hardware
//
#include <hip/hip_runtime.h>
#include <hip/hip_bf16.h>
#include <math.h>

#define WAVE 32
#define DFEAT 128
#define LLAYERS 3

typedef __attribute__((ext_vector_type(16))) __bf16 bf16x16;
typedef __attribute__((ext_vector_type(8)))  float  f32x8;

__device__ __forceinline__ float clip50(float v) { return fminf(fmaxf(v, -50.f), 50.f); }
__device__ __forceinline__ float n2n(float v) {
    if (isnan(v)) return 0.f;
    if (isinf(v)) return v > 0.f ? 1.f : 0.f;
    return v;
}

// ---------------------------------------------------------------- converts
__global__ void f32_to_bf16_kernel(const float* __restrict__ in, __bf16* __restrict__ out, int n) {
    int i = blockIdx.x * blockDim.x + threadIdx.x;
    if (i < n) out[i] = (__bf16)in[i];
}

__global__ void zero_kernel(float* __restrict__ p, int n) {
    int i = blockIdx.x * blockDim.x + threadIdx.x;
    if (i < n) p[i] = 0.f;
}

// ---------------------------------------------------------------- W pre-pack
// W[Kdim x 128] fp32 -> bf16 pairs in WMMA B-fragment order:
//   uint index = ((tile*nkb + kb)*32 + lane)*8 + v   (lane fragment = 32B run)
// k-map: k = kb*32 + (v<4 ? 2v : 16+2(v-4)) + 8*(lane>>4)
__global__ void pack_w_kernel(const float* __restrict__ W, unsigned* __restrict__ Wp, int Kdim) {
    int nkb = Kdim >> 5;
    int total = 8 * nkb * 32 * 8;
    int i = blockIdx.x * blockDim.x + threadIdx.x;
    if (i >= total) return;
    int v    = i & 7;
    int lane = (i >> 3) & 31;
    int rest = i >> 8;               // tile*nkb + kb
    int kb   = rest % nkb;
    int t    = rest / nkb;
    int half = lane >> 4;
    int nn   = lane & 15;
    int k    = kb * 32 + ((v < 4) ? (2 * v) : (16 + 2 * (v - 4))) + 8 * half;
    int col  = t * 16 + nn;
    __bf16 lo = (__bf16)W[(size_t)k * DFEAT + col];
    __bf16 hi = (__bf16)W[(size_t)(k + 1) * DFEAT + col];
    unsigned u = ((unsigned)__builtin_bit_cast(unsigned short, hi) << 16)
               |  (unsigned)__builtin_bit_cast(unsigned short, lo);
    Wp[i] = u;
}

// ---------------------------------------------------------------- fused QKVR WMMA GEMM
// For w in {q,k,v,r}: Y_w[n x 128] = clip(X @ W_w, +-50).
// X tile (16 rows) staged ONCE into LDS in A-fragment order via the CDNA5
// async global->LDS path, then reused for all four weight matrices.
// Block = 256 threads = 8 waves; wave w -> output cols [16w, 16w+16).
__global__ __launch_bounds__(256) void gemm_qkvr_wmma_kernel(
        const __bf16* __restrict__ X, const unsigned* __restrict__ Wp, size_t wStride,
        float* __restrict__ Y, size_t yStride, int n, int Kdim)
{
    __shared__ unsigned xs[(DFEAT / 32) * 32 * 8];   // nkb*32 lanes*8 uints, max 4KB
    const int rowBase = blockIdx.x * 16;
    const int tid  = threadIdx.x;
    const int nkb  = Kdim >> 5;

    if (rowBase + 16 > n) {                          // partial tile: pre-zero
        for (int i = tid; i < nkb * 256; i += blockDim.x) xs[i] = 0u;
        __syncthreads();
    }

    // async swizzled staging: one 16B chunk = (row, k0..k0+7)
    const int chunks = 16 * (Kdim >> 3);
    for (int c = tid; c < chunks; c += blockDim.x) {
        int row = c / (Kdim >> 3);
        int qd  = c - row * (Kdim >> 3);
        int k0  = qd * 8;
        int kb  = k0 >> 5;
        int rem = k0 & 31;
        int half  = (rem >> 3) & 1;
        int vbase = (rem & 16) ? 4 : 0;
        int fl    = row + 16 * half;                 // fragment lane
        int grow  = rowBase + row;
        if (grow < n) {
            unsigned ldsAddr =
                (unsigned)(uintptr_t)(void*)&xs[((kb * 32 + fl) * 8 + vbase)];
            unsigned long long ga =
                (unsigned long long)(uintptr_t)(X + (size_t)grow * Kdim + k0);
            asm volatile("global_load_async_to_lds_b128 %0, %1, off"
                         :: "v"(ldsAddr), "v"(ga) : "memory");
        }
    }
    asm volatile("s_wait_asynccnt 0x0" ::: "memory");
    __syncthreads();

    const int wave = tid / WAVE;
    const int lane = tid % WAVE;
    const int half = lane >> 4;
    const int l16  = lane & 15;
    const size_t fragOff = ((size_t)wave * nkb * 32 + lane) * 8;

#pragma unroll
    for (int w = 0; w < 4; ++w) {
        const unsigned* wp = Wp + w * wStride + fragOff;
        __builtin_prefetch(wp, 0, 1);                // global_prefetch_b8
        f32x8 acc = {};
        for (int kb = 0; kb < nkb; ++kb) {
            bf16x16 a = *(const bf16x16*)&xs[(kb * 32 + lane) * 8];   // 2x ds_load_b128
            bf16x16 b = *(const bf16x16*)(wp + (size_t)kb * 32 * 8);  // 2x global_load_b128
            acc = __builtin_amdgcn_wmma_f32_16x16x32_bf16(false, a, false, b,
                                                          (short)0, acc, false, false);
        }
        float* Yw = Y + w * yStride;
#pragma unroll
        for (int r = 0; r < 8; ++r) {
            int row = rowBase + half * 8 + r;        // C/D: VGPR r -> M=r (+8 hi lanes)
            if (row < n) Yw[(size_t)row * DFEAT + wave * 16 + l16] = clip50(acc[r]);
        }
    }
}

// ---------------------------------------------------------------- edge phase
__global__ void edge_alpha_kernel(const float* __restrict__ q, const float* __restrict__ k,
                                  const int* __restrict__ src, const int* __restrict__ dst,
                                  float* __restrict__ alpha, unsigned* __restrict__ amaxKey,
                                  int E_, int heads, int dh, int Dch, float scale)
{
    int idx = blockIdx.x * blockDim.x + threadIdx.x;
    if (idx >= E_ * heads) return;
    int e = idx / heads, h = idx - e * heads;
    int s = src[e], d = dst[e];
    const float* qp = q + (size_t)d * Dch + h * dh;
    const float* kp = k + (size_t)s * Dch + h * dh;
    float acc = 0.f;
    for (int i = 0; i < dh; i += 4) {                // 16B vector gathers
        float4 qv = *(const float4*)(qp + i);
        float4 kv = *(const float4*)(kp + i);
        acc += qv.x * kv.x + qv.y * kv.y + qv.z * kv.z + qv.w * kv.w;
    }
    float a = clip50(acc * scale);
    alpha[idx] = a;
    unsigned bits = __float_as_uint(a);
    unsigned key = (bits & 0x80000000u) ? ~bits : (bits | 0x80000000u);
    atomicMax(&amaxKey[(size_t)d * heads + h], key);
}

__global__ void edge_exp_kernel(const int* __restrict__ dst,
                                float* __restrict__ alpha, const unsigned* __restrict__ amaxKey,
                                float* __restrict__ denom, int E_, int heads)
{
    int idx = blockIdx.x * blockDim.x + threadIdx.x;
    if (idx >= E_ * heads) return;
    int e = idx / heads, h = idx - e * heads;
    int d = dst[e];
    unsigned key = amaxKey[(size_t)d * heads + h];
    float amax = (key & 0x80000000u) ? __uint_as_float(key & 0x7fffffffu)
                                     : __uint_as_float(~key);
    float ex = __expf(alpha[idx] - amax);
    alpha[idx] = ex;
    atomicAdd(&denom[(size_t)d * heads + h], ex);
}

// one thread per (edge, channel): a wave's 32 atomics land on 32 consecutive
// floats of the destination row -> coalesced atomic bursts into L2
__global__ void edge_msg_kernel(const float* __restrict__ v,
                                const int* __restrict__ src, const int* __restrict__ dst,
                                const float* __restrict__ alpha, const float* __restrict__ denom,
                                float* __restrict__ outAcc, int E_, int heads, int dh, int Dch)
{
    int idx = blockIdx.x * blockDim.x + threadIdx.x;
    if (idx >= E_ * Dch) return;
    int e = idx / Dch, c = idx - e * Dch;
    int h = c / dh;
    int s = src[e], d = dst[e];
    float a = n2n(alpha[(size_t)e * heads + h] /
                  (denom[(size_t)d * heads + h] + 1e-16f));
    float m = clip50(v[(size_t)s * Dch + c] * a);
    atomicAdd(&outAcc[(size_t)d * Dch + c], m);
}

__global__ void post_conv_kernel(const float* __restrict__ acc, const float* __restrict__ r,
                                 float* __restrict__ out, int total)
{
    int i = blockIdx.x * blockDim.x + threadIdx.x;
    if (i < total) out[i] = n2n(acc[i]) + r[i];
}

// ---------------------------------------------------------------- LN + ReLU
__global__ __launch_bounds__(256) void ln_relu_kernel(float* __restrict__ x,
                                                      const float* __restrict__ g,
                                                      const float* __restrict__ b, int n)
{
    int row = blockIdx.x * (blockDim.x / WAVE) + threadIdx.x / WAVE;
    int lane = threadIdx.x % WAVE;
    if (row >= n) return;
    float* xp = x + (size_t)row * DFEAT;
    float v[4];
    float s = 0.f;
#pragma unroll
    for (int i = 0; i < 4; ++i) { v[i] = xp[lane + 32 * i]; s += v[i]; }
#pragma unroll
    for (int m = 16; m; m >>= 1) s += __shfl_xor(s, m, WAVE);
    float mu = s * (1.f / DFEAT);
    float sq = 0.f;
#pragma unroll
    for (int i = 0; i < 4; ++i) { float d = v[i] - mu; sq += d * d; }
#pragma unroll
    for (int m = 16; m; m >>= 1) sq += __shfl_xor(sq, m, WAVE);
    float rstd = rsqrtf(sq * (1.f / DFEAT) + 1e-5f);
#pragma unroll
    for (int i = 0; i < 4; ++i) {
        int c = lane + 32 * i;
        float y = (v[i] - mu) * rstd * g[c] + b[c];
        xp[c] = fmaxf(y, 0.f);
    }
}

// ---------------------------------------------------------------- BatchNorm
__global__ __launch_bounds__(256) void bn_stats_kernel(const float* __restrict__ x,
                                                       float* __restrict__ colSum,
                                                       float* __restrict__ colSq, int n)
{
    int c   = threadIdx.x & (DFEAT - 1);
    int sub = threadIdx.x >> 7;
    float s = 0.f, q = 0.f;
    for (int r = blockIdx.x * 2 + sub; r < n; r += gridDim.x * 2) {
        float v = x[(size_t)r * DFEAT + c];
        s += v; q += v * v;
    }
    atomicAdd(&colSum[c], s);
    atomicAdd(&colSq[c], q);
}

__global__ void bn_apply_kernel(const float* __restrict__ x,
                                const float* __restrict__ colSum, const float* __restrict__ colSq,
                                const float* __restrict__ g, const float* __restrict__ b,
                                const float* __restrict__ identity,
                                float* __restrict__ hOut, __bf16* __restrict__ hOutB,
                                int n, int residual)
{
    int i = blockIdx.x * blockDim.x + threadIdx.x;
    if (i >= n * DFEAT) return;
    int c = i & (DFEAT - 1);
    float inv = 1.f / (float)n;
    float mu  = colSum[c] * inv;
    float var = colSq[c] * inv - mu * mu;
    float y = (x[i] - mu) * rsqrtf(var + 1e-5f) * g[c] + b[c];
    if (residual) y = clip50(y + identity[i]);
    hOut[i]  = y;
    hOutB[i] = (__bf16)y;
}

// ---------------------------------------------------------------- final tiny GEMM
__global__ void gemm_small_kernel(const float* __restrict__ X, const float* __restrict__ W,
                                  float* __restrict__ Y, int n, int Kdim, int Cc)
{
    int row = blockIdx.x * blockDim.x + threadIdx.x;
    if (row >= n) return;
    float acc[4] = {0.f, 0.f, 0.f, 0.f};
    const float* xp = X + (size_t)row * Kdim;
    for (int k = 0; k < Kdim; ++k) {
        float xv = xp[k];
        for (int c = 0; c < Cc; ++c) acc[c] += xv * W[k * Cc + c];
    }
    for (int c = 0; c < Cc; ++c) Y[(size_t)row * Cc + c] = clip50(acc[c]);
}

// ================================================================ host
extern "C" void kernel_launch(void* const* d_in, const int* in_sizes, int n_in,
                              void* d_out, int out_size, void* d_ws, size_t ws_size,
                              hipStream_t stream)
{
    const float* x     = (const float*)d_in[0];
    const float* pW[4] = {(const float*)d_in[1], (const float*)d_in[2],
                          (const float*)d_in[3], (const float*)d_in[4]};
    const float* ln0_g = (const float*)d_in[5];
    const float* ln0_b = (const float*)d_in[6];
    const float* bn0_g = (const float*)d_in[7];
    const float* bn0_b = (const float*)d_in[8];
    const float* cW[4] = {(const float*)d_in[9],  (const float*)d_in[10],
                          (const float*)d_in[11], (const float*)d_in[12]};
    const float* lns_g = (const float*)d_in[13];
    const float* lns_b = (const float*)d_in[14];
    const float* bns_g = (const float*)d_in[15];
    const float* bns_b = (const float*)d_in[16];
    const float* fW[4] = {(const float*)d_in[17], (const float*)d_in[18],
                          (const float*)d_in[19], (const float*)d_in[20]};
    const int*   ei    = (const int*)d_in[21];

    const int N_   = in_sizes[0] / 32;       // 50000
    const int E_   = in_sizes[21] / 2;       // 600000
    const int DIN_ = 32;
    const int* src = ei;
    const int* dst = ei + E_;

    // -------- workspace carve-out
    char* ws = (char*)d_ws;
    auto alloc = [&](size_t bytes) -> void* {
        void* p = (void*)ws;
        ws += (bytes + 255) & ~(size_t)255;
        return p;
    };
    const size_t packIn = 8 * (DIN_ / 32) * 32 * 8;    // uints per packed input W
    const size_t packD  = 8 * (DFEAT / 32) * 32 * 8;   // uints per packed DxD W
    const size_t NV     = (size_t)N_ * DFEAT;
    __bf16*   xb    = (__bf16*)alloc((size_t)N_ * DIN_ * 2);
    unsigned* pPack = (unsigned*)alloc(4 * packIn * 4);            // [w] contiguous
    unsigned* cPack = (unsigned*)alloc((size_t)LLAYERS * 4 * packD * 4); // [l][w]
    float* qkvr = (float*)alloc(4 * NV * 4);                       // q,k,v,r contiguous
    float* qf = qkvr, *kf = qkvr + NV, *vf = qkvr + 2 * NV, *rf = qkvr + 3 * NV;
    float* acc  = (float*)alloc(NV * 4);
    float* hA   = (float*)alloc(NV * 4);
    __bf16* hAb = (__bf16*)alloc(NV * 2);
    float*    alpha   = (float*)alloc((size_t)E_ * 4 * 4);
    unsigned* amaxKey = (unsigned*)alloc((size_t)N_ * 4 * 4);
    float*    denom   = (float*)alloc((size_t)N_ * 4 * 4);
    float*    colSum  = (float*)alloc(DFEAT * 4);
    float*    colSq   = (float*)alloc(DFEAT * 4);

    const int TB = 256;
    auto cdiv = [](int a, int b) { return (a + b - 1) / b; };
    auto zero = [&](float* p, int n) {
        zero_kernel<<<cdiv(n, TB), TB, 0, stream>>>(p, n);
    };

    // -------- conversions / weight packing
    f32_to_bf16_kernel<<<cdiv(N_ * DIN_, TB), TB, 0, stream>>>(x, xb, N_ * DIN_);
    for (int i = 0; i < 4; ++i)
        pack_w_kernel<<<cdiv((int)packIn, TB), TB, 0, stream>>>(pW[i], pPack + i * packIn, DIN_);
    for (int l = 0; l < LLAYERS; ++l)
        for (int i = 0; i < 4; ++i)
            pack_w_kernel<<<cdiv((int)packD, TB), TB, 0, stream>>>(
                cW[i] + (size_t)l * DFEAT * DFEAT, cPack + ((size_t)l * 4 + i) * packD, DFEAT);

    const float scale4 = 0.17677669529663687f;   // 32^-0.5
    const int rowTiles = cdiv(N_, 16);

    auto run_edges = [&](const float* q_, const float* k_, const float* v_, float* acc_,
                         int heads, int dh, int Dch, float scl) {
        zero((float*)amaxKey, N_ * heads);
        zero(denom, N_ * heads);
        zero(acc_, N_ * Dch);
        int eh = E_ * heads;
        edge_alpha_kernel<<<cdiv(eh, TB), TB, 0, stream>>>(q_, k_, src, dst, alpha, amaxKey,
                                                           E_, heads, dh, Dch, scl);
        edge_exp_kernel<<<cdiv(eh, TB), TB, 0, stream>>>(dst, alpha, amaxKey, denom, E_, heads);
        edge_msg_kernel<<<cdiv(E_ * Dch, TB), TB, 0, stream>>>(v_, src, dst, alpha, denom, acc_,
                                                               E_, heads, dh, Dch);
    };

    auto run_bn = [&](const float* xin, const float* g, const float* b,
                      const float* ident, int residual) {
        zero(colSum, DFEAT); zero(colSq, DFEAT);
        bn_stats_kernel<<<512, TB, 0, stream>>>(xin, colSum, colSq, N_);
        bn_apply_kernel<<<cdiv((int)NV, TB), TB, 0, stream>>>(
            xin, colSum, colSq, g, b, ident, hA, hAb, N_, residual);
    };

    // ---------------- stage 0: input conv (DIN=32 -> D=128), fused q/k/v/r
    gemm_qkvr_wmma_kernel<<<rowTiles, TB, 0, stream>>>(xb, pPack, packIn, qkvr, NV, N_, DIN_);
    run_edges(qf, kf, vf, acc, 4, 32, DFEAT, scale4);
    post_conv_kernel<<<cdiv((int)NV, TB), TB, 0, stream>>>(acc, rf, acc, (int)NV);
    ln_relu_kernel<<<cdiv(N_, 8), TB, 0, stream>>>(acc, ln0_g, ln0_b, N_);
    run_bn(acc, bn0_g, bn0_b, hA, 0);

    // ---------------- stages 1..L: residual conv blocks (D -> D)
    for (int l = 0; l < LLAYERS; ++l) {
        gemm_qkvr_wmma_kernel<<<rowTiles, TB, 0, stream>>>(
            hAb, cPack + (size_t)l * 4 * packD, packD, qkvr, NV, N_, DFEAT);
        run_edges(qf, kf, vf, acc, 4, 32, DFEAT, scale4);
        post_conv_kernel<<<cdiv((int)NV, TB), TB, 0, stream>>>(acc, rf, acc, (int)NV);
        ln_relu_kernel<<<cdiv(N_, 8), TB, 0, stream>>>(acc, lns_g + l * DFEAT, lns_b + l * DFEAT, N_);
        run_bn(acc, bns_g + l * DFEAT, bns_b + l * DFEAT, hA, 1);
    }

    // ---------------- final conv: D=128 -> C=4, heads=1
    const int Cc = 4;
    gemm_small_kernel<<<cdiv(N_, TB), TB, 0, stream>>>(hA, fW[0], qf, N_, DFEAT, Cc);
    gemm_small_kernel<<<cdiv(N_, TB), TB, 0, stream>>>(hA, fW[1], kf, N_, DFEAT, Cc);
    gemm_small_kernel<<<cdiv(N_, TB), TB, 0, stream>>>(hA, fW[2], vf, N_, DFEAT, Cc);
    gemm_small_kernel<<<cdiv(N_, TB), TB, 0, stream>>>(hA, fW[3], rf, N_, DFEAT, Cc);
    run_edges(qf, kf, vf, acc, 1, 4, Cc, 0.5f);
    post_conv_kernel<<<cdiv(N_ * Cc, TB), TB, 0, stream>>>(acc, rf, (float*)d_out, N_ * Cc);
}
